// AR_Transcriber_88974542504097
// MI455X (gfx1250) — matrix operand order repaired and checked by reading, not compiled, not run on hardware
//
#include <hip/hip_runtime.h>
#include <hip/hip_bf16.h>
#include <math.h>

// ---------------- types & helpers ----------------
typedef __attribute__((ext_vector_type(16))) __bf16 v16bf;
typedef __attribute__((ext_vector_type(8)))  float  v8f;

#define C16V   48
#define TDIM   512
#define FIN    229
#define TOUT   510
#define LSTMH  768
#define KX0    960      // padded xt K (944 -> 960)
#define PREVW  192      // padded prev width (176 -> 192)
#define NWG    48
#define PROJN  448      // padded 440

__device__ __forceinline__ float sigf(float x) { return 1.0f / (1.0f + __expf(-x)); }

__device__ __forceinline__ v8f wmma_bf16(v16bf a, v16bf b, v8f c) {
  return __builtin_amdgcn_wmma_f32_16x16x32_bf16(false, a, false, b, (short)0, c, false, false);
}

// A operand (16x32 bf16): lane l -> row l&15; elements 0..7 = K[hi*8 .. hi*8+7],
// elements 8..15 = K[16+hi*8 ..]. Requires `lane` in scope.
#define LOAD_A_FRAG(frag, base, ld, k0) do {                                   \
    const __bf16* _p = (base) + (size_t)(lane & 15) * (size_t)(ld) + (size_t)(k0) + (size_t)(((lane) >> 4) << 3); \
    _Pragma("unroll") for (int _i = 0; _i < 8; ++_i) { (frag)[_i] = _p[_i]; (frag)[_i + 8] = _p[_i + 16]; } \
  } while (0)

// B operand (32x16 bf16, B[k][n] = W[n][k]): lane -> column n = lane&15;
// elements 0..7 = K[hi*8 .. hi*8+7], elements 8..15 = K[16+hi*8 ..], read from row n of W.
#define LOAD_B_FRAG(frag, base, ld, k0) do {                                   \
    const __bf16* _p = (base) + (size_t)(lane & 15) * (size_t)(ld) + (size_t)(k0) + (size_t)(((lane) >> 4) << 3); \
    _Pragma("unroll") for (int _i = 0; _i < 8; ++_i) { (frag)[_i] = _p[_i]; (frag)[_i + 8] = _p[_i + 16]; } \
  } while (0)

__device__ __forceinline__ void grid_sync(unsigned* cnt, unsigned target) {
  __threadfence();
  __syncthreads();
  if (threadIdx.x == 0) {
    atomicAdd(cnt, 1u);
    while (*(volatile unsigned*)cnt < target) { __builtin_amdgcn_s_sleep(8); }
  }
  __syncthreads();
  __threadfence();
}

// ---------------- prep kernels ----------------
__global__ void cvt_pad_bf16(const float* __restrict__ src, __bf16* __restrict__ dst,
                             int drows, int dld, int srows, int sld) {
  size_t idx = (size_t)blockIdx.x * blockDim.x + threadIdx.x;
  size_t tot = (size_t)drows * dld;
  if (idx >= tot) return;
  int r = (int)(idx / dld), c = (int)(idx % dld);
  float v = (r < srows && c < sld) ? src[(size_t)r * sld + c] : 0.0f;
  dst[idx] = (__bf16)v;
}

// conv weights (O,I,3,3) f32 -> [O][tap(9)][64ci] bf16, ci padded 48->64 with zeros
__global__ void cvt_convw(const float* __restrict__ src, __bf16* __restrict__ dst,
                          int O, int I) {
  size_t idx = (size_t)blockIdx.x * blockDim.x + threadIdx.x;
  size_t tot = (size_t)O * 9 * 64;
  if (idx >= tot) return;
  int o = (int)(idx / 576);
  int r = (int)(idx % 576);
  int tap = r / 64, ci = r % 64;
  float v = (ci < I) ? src[((size_t)o * I + ci) * 9 + tap] : 0.0f;
  dst[idx] = (__bf16)v;
}

__global__ void vec_add_f32(const float* __restrict__ a, const float* __restrict__ b,
                            float* __restrict__ o, int n) {
  int i = blockIdx.x * blockDim.x + threadIdx.x;
  if (i < n) o[i] = a[i] + b[i];
}

__global__ void pad_copy_f32(const float* __restrict__ src, float* __restrict__ dst,
                             int n, int npad) {
  int i = blockIdx.x * blockDim.x + threadIdx.x;
  if (i < npad) dst[i] = (i < n) ? src[i] : 0.0f;
}

__global__ void init_state(__bf16* __restrict__ st, int n, unsigned* __restrict__ cnt) {
  size_t i = (size_t)blockIdx.x * blockDim.x + threadIdx.x;
  if (i < (size_t)n) st[i] = (__bf16)0.0f;
  if (i == 0) *cnt = 0u;
}

// ---------------- conv1: direct f32 -> channel-last padded [514][244][64] ----------------
__global__ void conv1_cl(const float* __restrict__ mel, const float* __restrict__ w,
                         const float* __restrict__ cb, const float* __restrict__ bg,
                         const float* __restrict__ bb, const float* __restrict__ bm,
                         const float* __restrict__ bv, __bf16* __restrict__ in2) {
  const int HP = TDIM + 2, WP = 244;
  size_t idx = (size_t)blockIdx.x * blockDim.x + threadIdx.x;
  size_t tot = (size_t)HP * WP * 64;
  if (idx >= tot) return;
  int yp = (int)(idx / ((size_t)WP * 64));
  int rem = (int)(idx % ((size_t)WP * 64));
  int xp = rem / 64, c = rem % 64;
  float val = 0.0f;
  if (c < C16V && yp >= 1 && yp <= TDIM && xp >= 1 && xp <= FIN) {
    int y = yp - 1, x = xp - 1;
    float acc = cb[c];
    #pragma unroll
    for (int dy = 0; dy < 3; ++dy)
      #pragma unroll
      for (int dx = 0; dx < 3; ++dx) {
        int yy = y + dy - 1, xx = x + dx - 1;
        if (yy >= 0 && yy < TDIM && xx >= 0 && xx < FIN)
          acc += w[c * 9 + dy * 3 + dx] * mel[(size_t)yy * FIN + xx];
      }
    float s = bg[c] * rsqrtf(bv[c] + 1e-5f);
    val = fmaxf(acc * s + (bb[c] - bm[c] * s), 0.0f);
  }
  in2[idx] = (__bf16)val;
}

// ---------------- conv2/conv3: channel-last direct-tap WMMA (no im2col, no LDS) -----------
// input [*][inWp][64] bf16 (zero borders/pads); weights [Cout][9][64]; 3x3, Cin=48->64.
// 64 pixels per WG (4 waves x 16-pixel M-tile), padW multiple of 16 so tiles never cross rows.
template <int NT>
__global__ __launch_bounds__(128) void conv_direct(
    const __bf16* __restrict__ inpad, int inWp,
    const __bf16* __restrict__ wts, const float* __restrict__ cb,
    const float* __restrict__ bg, const float* __restrict__ bb,
    const float* __restrict__ bm, const float* __restrict__ bv,
    __bf16* __restrict__ outp, int outW, int padW) {
  const int tid = threadIdx.x, lane = tid & 31, wid = tid >> 5;
  const int mwave = blockIdx.x * 64 + wid * 16;
  const int y = mwave / padW, x0 = mwave % padW;
  const v8f vzero = {};
  v8f acc[NT];
  #pragma unroll
  for (int n = 0; n < NT; ++n) acc[n] = vzero;
  #pragma unroll
  for (int tap = 0; tap < 9; ++tap) {
    const int dy = tap / 3, dx = tap % 3;
    const __bf16* Ab = inpad + ((size_t)(y + dy) * inWp + (x0 + dx)) * 64;
    #pragma unroll
    for (int kk = 0; kk < 2; ++kk) {
      v16bf a;
      LOAD_A_FRAG(a, Ab, 64, kk * 32);
      #pragma unroll
      for (int n = 0; n < NT; ++n) {
        v16bf b;
        LOAD_B_FRAG(b, wts + (size_t)(n * 16) * 576 + tap * 64, 576, kk * 32);
        acc[n] = wmma_bf16(a, b, acc[n]);
      }
    }
  }
  const int j = lane & 15, hi = lane >> 4;
  #pragma unroll
  for (int n = 0; n < NT; ++n) {
    int co = n * 16 + j;
    float s = bg[co] * rsqrtf(bv[co] + 1e-5f);
    float tt = bb[co] - bm[co] * s;
    float bias = cb[co];
    #pragma unroll
    for (int v = 0; v < 8; ++v) {
      int x = x0 + hi * 8 + v;
      if (x < outW) {
        float val = fmaxf((acc[n][v] + bias) * s + tt, 0.0f);
        outp[((size_t)y * outW + x) * (NT * 16) + co] = (__bf16)val;
      }
    }
  }
}

// pool over W: s2cl [512][229][48] -> conv3 input [512][132][64] (x-pad 1 halo + tile pad)
__global__ void pool2_cl(const __bf16* __restrict__ s2cl, __bf16* __restrict__ in3) {
  size_t idx = (size_t)blockIdx.x * blockDim.x + threadIdx.x;
  size_t tot = (size_t)TDIM * 132 * 64;
  if (idx >= tot) return;
  int y = (int)(idx / ((size_t)132 * 64));
  int rem = (int)(idx % ((size_t)132 * 64));
  int xp = rem / 64, c = rem % 64;
  float v = 0.0f;
  if (c < C16V && xp >= 1 && xp <= 114) {
    int w = xp - 1;
    float a = (float)s2cl[((size_t)y * FIN + 2 * w) * C16V + c];
    float b = (float)s2cl[((size_t)y * FIN + 2 * w + 1) * C16V + c];
    v = fmaxf(a, b);
  }
  in3[idx] = (__bf16)v;
}

// pool3 + layout: s3cl [510][114][96] -> s3p [512][5472] (k = c*57+w), rows 510/511 zero
__global__ void pool3_cl(const __bf16* __restrict__ s3cl, __bf16* __restrict__ s3p) {
  size_t idx = (size_t)blockIdx.x * blockDim.x + threadIdx.x;
  size_t tot = (size_t)512 * 5472;
  if (idx >= tot) return;
  int t = (int)(idx / 5472);
  int k = (int)(idx % 5472);
  float v = 0.0f;
  if (t < TOUT) {
    int c = k / 57, w = k % 57;
    float a = (float)s3cl[((size_t)t * 114 + 2 * w) * 96 + c];
    float b = (float)s3cl[((size_t)t * 114 + 2 * w + 1) * 96 + c];
    v = fmaxf(a, b);
  }
  s3p[idx] = (__bf16)v;
}

// ---------------- FC GEMM: acoustic[t][768] = s3p[t][:] @ fcw.T + fcb ----------------
__global__ __launch_bounds__(256) void fc_kernel(const __bf16* __restrict__ s3p,
                                                 const __bf16* __restrict__ fw,
                                                 const float* __restrict__ fb,
                                                 __bf16* __restrict__ aco) {
  const int tid = threadIdx.x, lane = tid & 31, wid = tid >> 5;
  const int mt = blockIdx.x;  // 0..31 -> rows t = mt*16..mt*16+15
  const __bf16* Ab = s3p + (size_t)mt * 16 * 5472;
  const v8f vzero = {};
  v8f acc[6];
  #pragma unroll
  for (int n = 0; n < 6; ++n) acc[n] = vzero;
  for (int kk = 0; kk < 171; ++kk) {
    v16bf a;
    LOAD_A_FRAG(a, Ab, 5472, kk * 32);
    __builtin_prefetch(fw + (size_t)((wid * 6) * 16) * 5472 + (size_t)(kk + 8) * 32, 0, 1);
    #pragma unroll
    for (int n = 0; n < 6; ++n) {
      v16bf b;
      LOAD_B_FRAG(b, fw + (size_t)((wid * 6 + n) * 16) * 5472, 5472, kk * 32);
      acc[n] = wmma_bf16(a, b, acc[n]);
    }
  }
  const int j = lane & 15, hi = lane >> 4;
  #pragma unroll
  for (int n = 0; n < 6; ++n) {
    int col = (wid * 6 + n) * 16 + j;
    float bias = fb[col];
    #pragma unroll
    for (int v = 0; v < 8; ++v) {
      int trow = mt * 16 + hi * 8 + v;
      aco[(size_t)trow * LSTMH + col] = (__bf16)(acc[n][v] + bias);
    }
  }
}

// ---------------- persistent autoregressive LSTM ----------------
// 48 WGs x 256 thr. WG owns 16 h-columns; 4 gate N-tiles x 2 K-half waves.
__global__ __launch_bounds__(256) void lstm_kernel(
    const __bf16* __restrict__ aco,                       // [16][512][768]
    const __bf16* __restrict__ W0, const __bf16* __restrict__ W0h,
    const __bf16* __restrict__ W1, const __bf16* __restrict__ W1h,
    const __bf16* __restrict__ PWb,                       // [448][768]
    const float* __restrict__ b0, const float* __restrict__ b1,
    const float* __restrict__ pbp, const float* __restrict__ emb,
    __bf16* __restrict__ h1g, __bf16* __restrict__ h2g,   // [2][16][768]
    __bf16* __restrict__ pvg,                             // [2][16][192]
    float* __restrict__ logits,                           // [16][448]
    unsigned* __restrict__ cnt, float* __restrict__ out) {
  __shared__ float zl[8][256];
  __shared__ float c1s[256], c2s[256];
  const int tid = threadIdx.x, lane = tid & 31, wid = tid >> 5;
  const int nt = wid & 3, kh = wid >> 2;
  const int wg = blockIdx.x;
  c1s[tid] = 0.0f;
  c2s[tid] = 0.0f;
  unsigned epoch = 0;
  const int HS = 16 * LSTMH;
  const int PS = 16 * PREVW;

  for (int t = 0; t < TOUT; ++t) {
    const int rd = t & 1, wr = rd ^ 1;
    // ---- layer 0: z = xt @ Wih0.T + h1 @ Whh0.T ----
    {
      const int ncol = nt * LSTMH + wg * 16;
      v8f acc = {};
      if (kh == 0) {
        const __bf16* Bw = W0 + (size_t)ncol * KX0;
        const __bf16* Aa = aco + (size_t)t * LSTMH;  // row stride = 512*768 (per batch)
        for (int kk = 0; kk < 24; ++kk) {
          v16bf a, b;
          LOAD_A_FRAG(a, Aa, (size_t)TDIM * LSTMH, kk * 32);
          LOAD_B_FRAG(b, Bw, KX0, kk * 32);
          acc = wmma_bf16(a, b, acc);
        }
        const __bf16* Ap = pvg + (size_t)rd * PS;
        for (int kk = 0; kk < 6; ++kk) {
          v16bf a, b;
          LOAD_A_FRAG(a, Ap, PREVW, kk * 32);
          LOAD_B_FRAG(b, Bw, KX0, 768 + kk * 32);
          acc = wmma_bf16(a, b, acc);
        }
      } else {
        const __bf16* Bw = W0h + (size_t)ncol * LSTMH;
        const __bf16* Ah = h1g + (size_t)rd * HS;
        for (int kk = 0; kk < 24; ++kk) {
          v16bf a, b;
          LOAD_A_FRAG(a, Ah, LSTMH, kk * 32);
          LOAD_B_FRAG(b, Bw, LSTMH, kk * 32);
          acc = wmma_bf16(a, b, acc);
        }
      }
      const int n = lane & 15, mb = (lane >> 4) * 8;
      #pragma unroll
      for (int v = 0; v < 8; ++v) zl[wid][(mb + v) * 16 + n] = acc[v];
    }
    __syncthreads();
    {  // gate combine layer 1 state
      const int j = tid & 15, col = wg * 16 + j;
      float zi = zl[0][tid] + zl[4][tid] + b0[col];
      float zf = zl[1][tid] + zl[5][tid] + b0[LSTMH + col];
      float zg = zl[2][tid] + zl[6][tid] + b0[2 * LSTMH + col];
      float zo = zl[3][tid] + zl[7][tid] + b0[3 * LSTMH + col];
      float c = sigf(zf) * c1s[tid] + sigf(zi) * tanhf(zg);
      c1s[tid] = c;
      float h = sigf(zo) * tanhf(c);
      h1g[(size_t)wr * HS + (tid >> 4) * LSTMH + col] = (__bf16)h;
    }
    grid_sync(cnt, ++epoch * NWG);  // h1(new) visible

    // ---- layer 1: z = h1 @ Wih1.T + h2 @ Whh1.T ----
    {
      const int ncol = nt * LSTMH + wg * 16;
      v8f acc = {};
      const __bf16* Bw = (kh == 0) ? (W1 + (size_t)ncol * LSTMH) : (W1h + (size_t)ncol * LSTMH);
      const __bf16* Ah = (kh == 0) ? (h1g + (size_t)wr * HS) : (h2g + (size_t)rd * HS);
      for (int kk = 0; kk < 24; ++kk) {
        v16bf a, b;
        LOAD_A_FRAG(a, Ah, LSTMH, kk * 32);
        LOAD_B_FRAG(b, Bw, LSTMH, kk * 32);
        acc = wmma_bf16(a, b, acc);
      }
      const int n = lane & 15, mb = (lane >> 4) * 8;
      #pragma unroll
      for (int v = 0; v < 8; ++v) zl[wid][(mb + v) * 16 + n] = acc[v];
    }
    __syncthreads();
    {  // gate combine layer 2 state
      const int j = tid & 15, col = wg * 16 + j;
      float zi = zl[0][tid] + zl[4][tid] + b1[col];
      float zf = zl[1][tid] + zl[5][tid] + b1[LSTMH + col];
      float zg = zl[2][tid] + zl[6][tid] + b1[2 * LSTMH + col];
      float zo = zl[3][tid] + zl[7][tid] + b1[3 * LSTMH + col];
      float c = sigf(zf) * c2s[tid] + sigf(zi) * tanhf(zg);
      c2s[tid] = c;
      float h = sigf(zo) * tanhf(c);
      h2g[(size_t)wr * HS + (tid >> 4) * LSTMH + col] = (__bf16)h;
    }
    grid_sync(cnt, ++epoch * NWG);  // h2(new) visible

    // ---- projection: logits = h2 @ pw.T + pb (WGs 0..27, 16 cols each) ----
    if (wg < 28) {
      const int n0 = wg * 16;
      v8f acc = {};
      const __bf16* Bw = PWb + (size_t)n0 * LSTMH;
      const __bf16* Ah = h2g + (size_t)wr * HS;
      for (int kk = wid * 3; kk < wid * 3 + 3; ++kk) {
        v16bf a, b;
        LOAD_A_FRAG(a, Ah, LSTMH, kk * 32);
        LOAD_B_FRAG(b, Bw, LSTMH, kk * 32);
        acc = wmma_bf16(a, b, acc);
      }
      const int n = lane & 15, mb = (lane >> 4) * 8;
      #pragma unroll
      for (int v = 0; v < 8; ++v) zl[wid][(mb + v) * 16 + n] = acc[v];
    }
    __syncthreads();
    if (wg < 28) {
      const int j = tid & 15, m = tid >> 4;
      float z = pbp[wg * 16 + j];
      #pragma unroll
      for (int w = 0; w < 8; ++w) z += zl[w][tid];
      logits[(size_t)m * PROJN + wg * 16 + j] = z;
    }
    grid_sync(cnt, ++epoch * NWG);  // logits visible

    // ---- argmax + embedding feedback + output store ----
    {
      int id = wg * 256 + tid;
      if (id < 16 * 88) {
        int m = id / 88, p = id - m * 88;
        const float* lg = logits + (size_t)m * PROJN + p * 5;
        float best = lg[0];
        int bi = 0;
        #pragma unroll
        for (int c = 1; c < 5; ++c)
          if (lg[c] > best) { best = lg[c]; bi = c; }
        float* o = out + (((size_t)m * TOUT + t) * 88 + p) * 5;
        #pragma unroll
        for (int c = 0; c < 5; ++c) o[c] = lg[c];
        pvg[(size_t)wr * PS + m * PREVW + p * 2]     = (__bf16)emb[bi * 2];
        pvg[(size_t)wr * PS + m * PREVW + p * 2 + 1] = (__bf16)emb[bi * 2 + 1];
      }
    }
    grid_sync(cnt, ++epoch * NWG);  // prev(new) visible
  }
}

// ---------------- host launcher ----------------
extern "C" void kernel_launch(void* const* d_in, const int* in_sizes, int n_in,
                              void* d_out, int out_size, void* d_ws, size_t ws_size,
                              hipStream_t stream) {
  (void)in_sizes; (void)n_in; (void)out_size; (void)ws_size;
  const float* mel  = (const float*)d_in[0];
  const float* c1w  = (const float*)d_in[1];
  const float* c1b  = (const float*)d_in[2];
  const float* c2w  = (const float*)d_in[3];
  const float* c2b  = (const float*)d_in[4];
  const float* c3w  = (const float*)d_in[5];
  const float* c3b  = (const float*)d_in[6];
  const float* bn1g = (const float*)d_in[7];
  const float* bn1b = (const float*)d_in[8];
  const float* bn1m = (const float*)d_in[9];
  const float* bn1v = (const float*)d_in[10];
  const float* bn2g = (const float*)d_in[11];
  const float* bn2b = (const float*)d_in[12];
  const float* bn2m = (const float*)d_in[13];
  const float* bn2v = (const float*)d_in[14];
  const float* bn3g = (const float*)d_in[15];
  const float* bn3b = (const float*)d_in[16];
  const float* bn3m = (const float*)d_in[17];
  const float* bn3v = (const float*)d_in[18];
  const float* fcw  = (const float*)d_in[19];
  const float* fcb  = (const float*)d_in[20];
  const float* wih0 = (const float*)d_in[21];
  const float* whh0 = (const float*)d_in[22];
  const float* bih0 = (const float*)d_in[23];
  const float* bhh0 = (const float*)d_in[24];
  const float* wih1 = (const float*)d_in[25];
  const float* whh1 = (const float*)d_in[26];
  const float* bih1 = (const float*)d_in[27];
  const float* bhh1 = (const float*)d_in[28];
  const float* pw   = (const float*)d_in[29];
  const float* pb   = (const float*)d_in[30];
  const float* emb  = (const float*)d_in[31];
  float* outp = (float*)d_out;

  // ---- carve workspace ----
  char* base = (char*)d_ws;
  size_t off = 0;
  auto carve = [&](size_t bytes) -> char* {
    char* p = base + off;
    off = (off + bytes + 255) & ~(size_t)255;
    return p;
  };
  __bf16* W0   = (__bf16*)carve((size_t)3072 * 960 * 2);
  __bf16* W0h  = (__bf16*)carve((size_t)3072 * 768 * 2);
  __bf16* W1   = (__bf16*)carve((size_t)3072 * 768 * 2);
  __bf16* W1h  = (__bf16*)carve((size_t)3072 * 768 * 2);
  __bf16* PWb  = (__bf16*)carve((size_t)448 * 768 * 2);
  __bf16* FCWb = (__bf16*)carve((size_t)768 * 5472 * 2);
  __bf16* C2Wb = (__bf16*)carve((size_t)48 * 576 * 2);
  __bf16* C3Wb = (__bf16*)carve((size_t)96 * 576 * 2);
  float*  b0   = (float*)carve(3072 * 4);
  float*  b1   = (float*)carve(3072 * 4);
  float*  pbp  = (float*)carve(448 * 4);
  __bf16* aco  = (__bf16*)carve((size_t)16 * 512 * 768 * 2);
  __bf16* in2  = (__bf16*)carve((size_t)514 * 244 * 64 * 2);   // conv2 input, channel-last
  __bf16* s2cl = (__bf16*)carve((size_t)512 * 229 * 48 * 2);   // conv2 output, channel-last
  __bf16* in3  = (__bf16*)carve((size_t)512 * 132 * 64 * 2);   // conv3 input, channel-last
  __bf16* s3cl = (__bf16*)carve((size_t)510 * 114 * 96 * 2);   // conv3 output, channel-last
  __bf16* s3p  = (__bf16*)carve((size_t)512 * 5472 * 2);
  __bf16* stateb = (__bf16*)carve((size_t)(2 * 16 * 768 * 2 + 2 * 16 * 192) * 2);
  __bf16* h1g = stateb;
  __bf16* h2g = stateb + 2 * 16 * 768;
  __bf16* pvg = stateb + 4 * 16 * 768;
  float*  logits = (float*)carve((size_t)16 * 448 * 4);
  unsigned* cnt  = (unsigned*)carve(256);

  auto g1 = [](size_t n) { return (unsigned)((n + 255) / 256); };

  // ---- weight conversion / bias prep (every call; deterministic) ----
  cvt_pad_bf16<<<g1((size_t)3072 * 960), 256, 0, stream>>>(wih0, W0, 3072, 960, 3072, 944);
  cvt_pad_bf16<<<g1((size_t)3072 * 768), 256, 0, stream>>>(whh0, W0h, 3072, 768, 3072, 768);
  cvt_pad_bf16<<<g1((size_t)3072 * 768), 256, 0, stream>>>(wih1, W1, 3072, 768, 3072, 768);
  cvt_pad_bf16<<<g1((size_t)3072 * 768), 256, 0, stream>>>(whh1, W1h, 3072, 768, 3072, 768);
  cvt_pad_bf16<<<g1((size_t)448 * 768), 256, 0, stream>>>(pw, PWb, 448, 768, 440, 768);
  cvt_pad_bf16<<<g1((size_t)768 * 5472), 256, 0, stream>>>(fcw, FCWb, 768, 5472, 768, 5472);
  cvt_convw<<<g1((size_t)48 * 576), 256, 0, stream>>>(c2w, C2Wb, 48, 48);
  cvt_convw<<<g1((size_t)96 * 576), 256, 0, stream>>>(c3w, C3Wb, 96, 48);
  vec_add_f32<<<g1(3072), 256, 0, stream>>>(bih0, bhh0, b0, 3072);
  vec_add_f32<<<g1(3072), 256, 0, stream>>>(bih1, bhh1, b1, 3072);
  pad_copy_f32<<<g1(448), 256, 0, stream>>>(pb, pbp, 440, 448);
  init_state<<<g1(2 * 16 * 768 * 2 + 2 * 16 * 192), 256, 0, stream>>>(
      stateb, 2 * 16 * 768 * 2 + 2 * 16 * 192, cnt);

  // ---- conv stack + FC, batch-serial to reuse scratch ----
  for (int b = 0; b < 16; ++b) {
    conv1_cl<<<g1((size_t)514 * 244 * 64), 256, 0, stream>>>(
        mel + (size_t)b * 512 * 229, c1w, c1b, bn1g, bn1b, bn1m, bn1v, in2);
    conv_direct<3><<<(512 * 240) / 64, 128, 0, stream>>>(
        in2, 244, C2Wb, c2b, bn2g, bn2b, bn2m, bn2v, s2cl, 229, 240);
    pool2_cl<<<g1((size_t)512 * 132 * 64), 256, 0, stream>>>(s2cl, in3);
    conv_direct<6><<<(510 * 128) / 64, 128, 0, stream>>>(
        in3, 132, C3Wb, c3b, bn3g, bn3b, bn3m, bn3v, s3cl, 114, 128);
    pool3_cl<<<g1((size_t)512 * 5472), 256, 0, stream>>>(s3cl, s3p);
    fc_kernel<<<32, 256, 0, stream>>>(s3p, FCWb, fcb, aco + (size_t)b * 512 * 768);
  }

  // ---- persistent autoregressive LSTM + projection + argmax feedback ----
  lstm_kernel<<<NWG, 256, 0, stream>>>(aco, W0, W0h, W1, W1h, PWb, b0, b1, pbp, emb,
                                       h1g, h2g, pvg, logits, cnt, outp);
}
